// TrainableExpertPool_7464653161201
// MI455X (gfx1250) — matrix operand order repaired and checked by reading, not compiled, not run on hardware
//
#include <hip/hip_runtime.h>
#include <hip/hip_bf16.h>

typedef unsigned short u16;
typedef unsigned int   u32;
typedef __attribute__((ext_vector_type(16))) __bf16 v16bf;
typedef __attribute__((ext_vector_type(16))) u16    v16u;
typedef __attribute__((ext_vector_type(8)))  float  v8f;
typedef int v4i __attribute__((vector_size(16)));   // param type of async-LDS builtin

#if defined(__HIP_DEVICE_COMPILE__) && __has_builtin(__builtin_amdgcn_global_load_async_to_lds_b128)
#define HAVE_ASYNC_LDS 1
#else
#define HAVE_ASYNC_LDS 0
#endif

#define TB 256        // 8 waves of 32
#define BLK_M 128     // block tile M (4 waves x 32 rows)
#define BLK_N 64      // block tile N (2 waves x 32 cols)
#define BK 32         // K chunk = one bf16 WMMA K step
#define LSTRIDE 40    // padded LDS row stride in u16 (80B: 16B-aligned, conflict-free)

// Problem dims (match setup_inputs)
#define B_ 8
#define S_ 512
#define H_ 512
#define I_ 2048
#define V_ 32000

// Native bf16 convert (RNE) -> v_cvt_*_bf16_f32, not bit-twiddle VALU
__device__ __forceinline__ u16 f32_to_bf16(float f) {
  __bf16 h = (__bf16)f;
  return __builtin_bit_cast(u16, h);
}

// Stage a ROWSx32 f32 tile (row-major, leading dim ld) into LDS as bf16.
// Pair-wise: float2 load -> packed bf16 -> one dword LDS store.
template <int ROWS>
__device__ __forceinline__ void stage_f32(const float* __restrict__ g, int ld,
                                          u16* Ls, int tid) {
  #pragma unroll
  for (int i = tid; i < ROWS * 16; i += TB) {
    int r = i >> 4, c2 = (i & 15) * 2;
    float2 f = *(const float2*)(g + (size_t)r * ld + c2);
    u32 v = (u32)f32_to_bf16(f.x) | ((u32)f32_to_bf16(f.y) << 16);
    *(u32*)(Ls + r * LSTRIDE + c2) = v;
  }
}

// Stage a ROWSx32 bf16 tile into LDS: async DMA (GLOBAL_LOAD_ASYNC_TO_LDS_B128)
// when available, else 16B vector copies.
template <int ROWS>
__device__ __forceinline__ void stage_bf16(const u16* __restrict__ g, int ld,
                                           u16* Ls, int tid) {
  #pragma unroll
  for (int i = tid; i < ROWS * 4; i += TB) {
    int r = i >> 2, c8 = (i & 3) * 8;  // 16-byte chunk
#if HAVE_ASYNC_LDS
    __builtin_amdgcn_global_load_async_to_lds_b128(
        (v4i*)(g + (size_t)r * ld + c8), (v4i*)(Ls + r * LSTRIDE + c8), 0, 0);
#else
    *(uint4*)(Ls + r * LSTRIDE + c8) = *(const uint4*)(g + (size_t)r * ld + c8);
#endif
  }
}

#if HAVE_ASYNC_LDS
__device__ __forceinline__ void wait_async() {
#if __has_builtin(__builtin_amdgcn_s_wait_asynccnt)
  __builtin_amdgcn_s_wait_asynccnt(0);
#else
  asm volatile("s_wait_asynccnt 0x0" ::: "memory");
#endif
}
#endif

// A fragment 16x32 bf16. Lanes 0-15: row M=lane, K in {0..7,16..23};
// lanes 16-31: row M=lane-16, K in {8..15,24..31}.
__device__ __forceinline__ v16bf frag_a(const u16* Ls, int row0, int lane) {
  const u16* p = Ls + (row0 + (lane & 15)) * LSTRIDE + ((lane < 16) ? 0 : 8);
  v16u t;
  #pragma unroll
  for (int e = 0; e < 8; ++e) { t[e] = p[e]; t[8 + e] = p[16 + e]; }
  return __builtin_bit_cast(v16bf, t);
}

// B fragment 32x16 bf16 with B[k][n] = Bs[n][k] (tile stored N-major, K contig).
// Lanes 0-15: K in {0..7,16..23}; lanes 16-31: K in {8..15,24..31}.
__device__ __forceinline__ v16bf frag_b(const u16* Ls, int col0, int lane) {
  const u16* p = Ls + (col0 + (lane & 15)) * LSTRIDE + ((lane < 16) ? 0 : 8);
  v16u t;
  #pragma unroll
  for (int e = 0; e < 8; ++e) { t[e] = p[e]; t[8 + e] = p[16 + e]; }
  return __builtin_bit_cast(v16bf, t);
}

__device__ __forceinline__ v8f wmma_bf16(v16bf a, v16bf b, v8f c) {
  return __builtin_amdgcn_wmma_f32_16x16x32_bf16(false, a, false, b, (short)0, c,
                                                 false, false);
}

// ---------------- kernel 1: top-2 expert selection ----------------
__global__ void topk_kernel(const float* __restrict__ ep,
                            int* __restrict__ ids, float* __restrict__ nprob) {
  int b = threadIdx.x;
  if (b >= B_) return;
  const float* p = ep + b * 16;
  int i1 = 0; float v1 = p[0];
  #pragma unroll
  for (int i = 1; i < 16; ++i) if (p[i] > v1) { v1 = p[i]; i1 = i; }
  int i2 = -1; float v2 = -1e30f;
  #pragma unroll
  for (int i = 0; i < 16; ++i) if (i != i1 && p[i] > v2) { v2 = p[i]; i2 = i; }
  float s = 1.f / (v1 + v2 + 1e-8f);
  ids[2 * b] = i1;   ids[2 * b + 1] = i2;
  nprob[2 * b] = v1 * s; nprob[2 * b + 1] = v2 * s;
}

// ---------------- kernel 2: fused gate/up + SiLU, act stored bf16 ----------------
__global__ void __launch_bounds__(TB)
moe_gateup_act(const float* __restrict__ hs, const float* __restrict__ Wg,
               const float* __restrict__ Wu, const int* __restrict__ ids,
               u16* __restrict__ act) {
  __shared__ u16 As[BLK_M * LSTRIDE];
  __shared__ u16 Gs[BLK_N * LSTRIDE];
  __shared__ u16 Us[BLK_N * LSTRIDE];
  const int tid = threadIdx.x, lane = tid & 31, wave = tid >> 5;
  const int wm = (wave & 3) * 32, wn = (wave >> 2) * 32;
  const int M0 = blockIdx.x * BLK_M, N0 = blockIdx.y * BLK_N;
  const int b = blockIdx.z >> 1, k = blockIdx.z & 1;
  const int e = ids[b * 2 + k];
  const float* A = hs + (size_t)b * S_ * H_ + (size_t)M0 * H_;
  const float* G = Wg + (size_t)e * I_ * H_ + (size_t)N0 * H_;
  const float* U = Wu + (size_t)e * I_ * H_ + (size_t)N0 * H_;
  v8f accg[2][2] = {}; v8f accu[2][2] = {};
  for (int kk = 0; kk < H_; kk += BK) {
    if (kk + BK < H_) {
      __builtin_prefetch(A + kk + BK, 0, 1);   // global_prefetch_b8
      __builtin_prefetch(G + kk + BK, 0, 1);
      __builtin_prefetch(U + kk + BK, 0, 1);
    }
    stage_f32<BLK_M>(A + kk, H_, As, tid);
    stage_f32<BLK_N>(G + kk, H_, Gs, tid);
    stage_f32<BLK_N>(U + kk, H_, Us, tid);
    __syncthreads();
    v16bf a0 = frag_a(As, wm, lane);
    v16bf a1 = frag_a(As, wm + 16, lane);
    #pragma unroll
    for (int ni = 0; ni < 2; ++ni) {
      v16bf bg = frag_b(Gs, wn + ni * 16, lane);
      accg[0][ni] = wmma_bf16(a0, bg, accg[0][ni]);
      accg[1][ni] = wmma_bf16(a1, bg, accg[1][ni]);
      v16bf bu = frag_b(Us, wn + ni * 16, lane);
      accu[0][ni] = wmma_bf16(a0, bu, accu[0][ni]);
      accu[1][ni] = wmma_bf16(a1, bu, accu[1][ni]);
    }
    __syncthreads();
  }
  u16* dst = act + (size_t)blockIdx.z * S_ * I_;
  const int mb = (lane < 16) ? 0 : 8, nc = lane & 15;
  #pragma unroll
  for (int mi = 0; mi < 2; ++mi)
    #pragma unroll
    for (int ni = 0; ni < 2; ++ni)
      #pragma unroll
      for (int r = 0; r < 8; ++r) {
        float g = accg[mi][ni][r], u = accu[mi][ni][r];
        float s = (g / (1.f + __expf(-g))) * u;   // SiLU(g) * u
        dst[(size_t)(M0 + wm + mi * 16 + mb + r) * I_ + (N0 + wn + ni * 16 + nc)] =
            f32_to_bf16(s);
      }
}

// ---------------- kernel 3: down-proj, both experts, prob-weighted ----------------
__global__ void __launch_bounds__(TB)
moe_down_combine(const u16* __restrict__ act, const float* __restrict__ Wd,
                 const int* __restrict__ ids, const float* __restrict__ probs,
                 float* __restrict__ comb) {
  __shared__ u16 As[BLK_M * LSTRIDE];
  __shared__ u16 Bs[BLK_N * LSTRIDE];
  const int tid = threadIdx.x, lane = tid & 31, wave = tid >> 5;
  const int wm = (wave & 3) * 32, wn = (wave >> 2) * 32;
  const int M0 = blockIdx.x * BLK_M, N0 = blockIdx.y * BLK_N;
  const int b = blockIdx.z;
  v8f res[2][2] = {};
  for (int k = 0; k < 2; ++k) {
    const int e = ids[b * 2 + k];
    const float p = probs[b * 2 + k];
    const u16* A = act + (size_t)(b * 2 + k) * S_ * I_ + (size_t)M0 * I_;
    const float* Bw = Wd + (size_t)e * H_ * I_ + (size_t)N0 * I_;
    v8f acc[2][2] = {};
    for (int kk = 0; kk < I_; kk += BK) {
      if (kk + BK < I_) {
        __builtin_prefetch(A + kk + BK, 0, 1);
        __builtin_prefetch(Bw + kk + BK, 0, 1);
      }
      stage_bf16<BLK_M>(A + kk, I_, As, tid);   // async DMA path when available
      stage_f32<BLK_N>(Bw + kk, I_, Bs, tid);
#if HAVE_ASYNC_LDS
      wait_async();                             // s_wait_asynccnt 0
#endif
      __syncthreads();
      v16bf a0 = frag_a(As, wm, lane);
      v16bf a1 = frag_a(As, wm + 16, lane);
      #pragma unroll
      for (int ni = 0; ni < 2; ++ni) {
        v16bf bb = frag_b(Bs, wn + ni * 16, lane);
        acc[0][ni] = wmma_bf16(a0, bb, acc[0][ni]);
        acc[1][ni] = wmma_bf16(a1, bb, acc[1][ni]);
      }
      __syncthreads();
    }
    #pragma unroll
    for (int mi = 0; mi < 2; ++mi)
      #pragma unroll
      for (int ni = 0; ni < 2; ++ni)
        #pragma unroll
        for (int r = 0; r < 8; ++r) res[mi][ni][r] += p * acc[mi][ni][r];
  }
  float* dst = comb + (size_t)b * S_ * H_;
  const int mb = (lane < 16) ? 0 : 8, nc = lane & 15;
  #pragma unroll
  for (int mi = 0; mi < 2; ++mi)
    #pragma unroll
    for (int ni = 0; ni < 2; ++ni)
      #pragma unroll
      for (int r = 0; r < 8; ++r)
        dst[(size_t)(M0 + wm + mi * 16 + mb + r) * H_ + (N0 + wn + ni * 16 + nc)] =
            res[mi][ni][r];
}

// ---------------- kernel 4: output projection ----------------
__global__ void __launch_bounds__(TB)
out_proj(const float* __restrict__ comb, const float* __restrict__ Wout,
         float* __restrict__ out) {
  __shared__ u16 As[BLK_M * LSTRIDE];
  __shared__ u16 Bs[BLK_N * LSTRIDE];
  const int tid = threadIdx.x, lane = tid & 31, wave = tid >> 5;
  const int wm = (wave & 3) * 32, wn = (wave >> 2) * 32;
  const int M0 = blockIdx.x * BLK_M, N0 = blockIdx.y * BLK_N;
  const int b = blockIdx.z;
  const float* A = comb + (size_t)b * S_ * H_ + (size_t)M0 * H_;
  const float* Bw = Wout + (size_t)N0 * H_;
  v8f acc[2][2] = {};
  for (int kk = 0; kk < H_; kk += BK) {
    if (kk + BK < H_) {
      __builtin_prefetch(A + kk + BK, 0, 1);
      __builtin_prefetch(Bw + kk + BK, 0, 1);
    }
    stage_f32<BLK_M>(A + kk, H_, As, tid);
    stage_f32<BLK_N>(Bw + kk, H_, Bs, tid);
    __syncthreads();
    v16bf a0 = frag_a(As, wm, lane);
    v16bf a1 = frag_a(As, wm + 16, lane);
    #pragma unroll
    for (int ni = 0; ni < 2; ++ni) {
      v16bf bb = frag_b(Bs, wn + ni * 16, lane);
      acc[0][ni] = wmma_bf16(a0, bb, acc[0][ni]);
      acc[1][ni] = wmma_bf16(a1, bb, acc[1][ni]);
    }
    __syncthreads();
  }
  float* dst = out + (size_t)b * S_ * V_;
  const int mb = (lane < 16) ? 0 : 8, nc = lane & 15;
  #pragma unroll
  for (int mi = 0; mi < 2; ++mi)
    #pragma unroll
    for (int ni = 0; ni < 2; ++ni)
      #pragma unroll
      for (int r = 0; r < 8; ++r)
        dst[(size_t)(M0 + wm + mi * 16 + mb + r) * V_ + (N0 + wn + ni * 16 + nc)] =
            acc[mi][ni][r];
}

extern "C" void kernel_launch(void* const* d_in, const int* in_sizes, int n_in,
                              void* d_out, int out_size, void* d_ws, size_t ws_size,
                              hipStream_t stream) {
  (void)in_sizes; (void)n_in; (void)out_size; (void)ws_size;
  const float* hs   = (const float*)d_in[0];   // (8,512,512)
  const float* ep   = (const float*)d_in[1];   // (8,16)
  const float* Wg   = (const float*)d_in[2];   // (16,2048,512)
  const float* Wu   = (const float*)d_in[3];   // (16,2048,512)
  const float* Wd   = (const float*)d_in[4];   // (16,512,2048)
  const float* Wout = (const float*)d_in[5];   // (32000,512)
  // d_in[6] = top_k (== 2 per setup; hardcoded)

  char* ws = (char*)d_ws;
  int*   ids  = (int*)ws;                                   // 16 ints
  float* prob = (float*)(ws + 64);                          // 16 floats
  u16*   act  = (u16*)(ws + 256);                           // 32 MiB bf16 acts
  float* comb = (float*)(ws + 256 + (size_t)B_ * 2 * S_ * I_ * sizeof(u16));
  float* out  = (float*)d_out;

  topk_kernel<<<1, 32, 0, stream>>>(ep, ids, prob);
  moe_gateup_act<<<dim3(S_ / BLK_M, I_ / BLK_N, B_ * 2), TB, 0, stream>>>(hs, Wg, Wu, ids, act);
  moe_down_combine<<<dim3(S_ / BLK_M, H_ / BLK_N, B_), TB, 0, stream>>>(act, Wd, ids, prob, comb);
  out_proj<<<dim3(S_ / BLK_M, V_ / BLK_N, B_), TB, 0, stream>>>(comb, Wout, out);
}
